// MessagePassing_22497038696556
// MI455X (gfx1250) — compile-verified, matrix operand order checked by reading, not run on hardware
//
#include <hip/hip_runtime.h>
#include <math.h>

// ---------------------------------------------------------------------------
// CDNA5 / gfx1250 implementation of the equivariant message-passing layer.
// Edge MLPs (the only compute-heavy part) run on v_wmma_f32_16x16x32_f16.
// Node-side ops and graph gather/scatter are scalar + f32 global atomics.
// ---------------------------------------------------------------------------

typedef __attribute__((ext_vector_type(16))) _Float16 v16h;
typedef __attribute__((ext_vector_type(8)))  float    v8f;

#define WAVES_PER_BLOCK 8   // 256 threads = 8 wave32

// Fast approximate math: single-instruction v_rcp_f32 / v_exp_f32 /
// v_sin_f32 / v_cos_f32 instead of IEEE division + libm argument reduction.
__device__ __forceinline__ float fast_rcp(float x) { return __builtin_amdgcn_rcpf(x); }
__device__ __forceinline__ float silu_f(float x) {
  return x * fast_rcp(1.0f + __expf(-x));
}
__device__ __forceinline__ float sigmoid_f(float x) {
  return fast_rcp(1.0f + __expf(-x));
}

__device__ __forceinline__ v8f wmma_f16(v16h a, v16h b, v8f c) {
  // D = A(16x32 f16) * B(32x16 f16) + C(16x16 f32)
  return __builtin_amdgcn_wmma_f32_16x16x32_f16(
      /*neg_a=*/false, a, /*neg_b=*/false, b,
      /*c_mod=*/(short)0, c, /*reuse_a=*/false, /*reuse_b=*/false);
}

// v16h element i <-> K index for the 16-bit 16x32 A/B operand layout:
//   VGPR j holds elements {2j,2j+1}; lanes 0-15 hold K 0-7 & 16-23,
//   lanes 16-31 hold K 8-15 & 24-31 (same rows/cols as lane&15).
__device__ __forceinline__ int kmap(int i, int hi) {
  return ((i & 8) << 1) + (hi << 3) + (i & 7);
}

// Load one 16-column B tile from a row-major f32 weight W[kreal x ncols],
// zero-padding K to kbase..kbase+31 and columns beyond ncols.
__device__ __forceinline__ v16h load_b(const float* __restrict__ W, int ncols,
                                       int kreal, int n0, int kbase,
                                       float scale, int lane) {
  const int n  = n0 + (lane & 15);
  const int hi = (lane >> 4) & 1;
  v16h b;
#pragma unroll
  for (int i = 0; i < 16; ++i) {
    const int k = kbase + kmap(i, hi);
    float v = 0.0f;
    if (k < kreal && n < ncols) v = W[k * ncols + n] * scale;
    b[i] = (_Float16)v;
  }
  return b;
}

// ---------------------------------------------------------------------------
__global__ void zero_kernel(float* __restrict__ p, int n) {
  int i = blockIdx.x * blockDim.x + threadIdx.x;
  if (i < n) p[i] = 0.0f;
}

// xf = (x @ lin1_w)/4 * a   (sc-part is recomputed in node_mid)
__global__ void node_pre_kernel(const float* __restrict__ x,
                                const float* __restrict__ attr,
                                const float* __restrict__ lin1_w, // 16x16
                                float* __restrict__ xf, int N) {
  int i = blockIdx.x * blockDim.x + threadIdx.x;
  if (i >= N) return;
  const float a = attr[i];
  float xs[16];
#pragma unroll
  for (int k = 0; k < 16; ++k) xs[k] = x[i * 16 + k];
#pragma unroll
  for (int j = 0; j < 16; ++j) {
    float s = 0.0f;
#pragma unroll
    for (int k = 0; k < 16; ++k) s += xs[k] * lin1_w[k * 16 + j];
    xf[i * 16 + j] = s * 0.25f * a;
  }
}

// ---------------------------------------------------------------------------
// Edge MLP #1 on WMMA + gather xf[src] + scatter-add into mid (N x 64).
// One wave handles a tile of 16 edges.
// ---------------------------------------------------------------------------
__global__ __launch_bounds__(256) void edge_mlp1_kernel(
    const float* __restrict__ es,      // E x 16 edge_scalars
    const float* __restrict__ eattr,   // E x 4  [sh0, sh1x, sh1y, sh1z]
    const int*   __restrict__ esrc,
    const int*   __restrict__ edst,
    const float* __restrict__ fc1_w1,  // 16x64
    const float* __restrict__ fc1_w2,  // 64x32
    const float* __restrict__ xf,      // N x 16
    float* __restrict__ mid,           // N x 64 (accumulated)
    int E) {
  __shared__ _Float16 hbuf[WAVES_PER_BLOCK * 16 * 64];  // 16 KB
  const int lane = threadIdx.x & 31;
  const int wave = threadIdx.x >> 5;
  const int hi   = (lane >> 4) & 1;
  const int lidx = lane & 15;
  _Float16* hb = &hbuf[wave * 16 * 64];

  // Resident B tiles: layer1 (fc1_w1/4), layer2 (fc1_w2/8)
  v16h B1[4], B2[2][2];
#pragma unroll
  for (int t = 0; t < 4; ++t) B1[t] = load_b(fc1_w1, 64, 16, t * 16, 0, 0.25f, lane);
#pragma unroll
  for (int t = 0; t < 2; ++t)
#pragma unroll
    for (int kc = 0; kc < 2; ++kc)
      B2[t][kc] = load_b(fc1_w2, 32, 64, t * 16, kc * 32, 0.125f, lane);

  const int ntiles = (E + 15) >> 4;
  const int stride = gridDim.x * WAVES_PER_BLOCK;
  for (int tile = blockIdx.x * WAVES_PER_BLOCK + wave; tile < ntiles; tile += stride) {
    const int e0 = tile << 4;

    // Edge metadata for edges e0..e0+15 (duplicated in both lane halves).
    const int  me    = e0 + lidx;
    const int  mec   = (me < E) ? me : (E - 1);
    const int  m_src = esrc[mec];
    const int  m_dst = edst[mec];
    const float m_sh0 = eattr[mec * 4 + 0];
    const float m_s1x = eattr[mec * 4 + 1];
    const float m_s1y = eattr[mec * 4 + 2];
    const float m_s1z = eattr[mec * 4 + 3];

    // A operand: 16 edges x 16 scalars (K padded to 32 with zeros).
    v16h a;
    {
      const int   ee   = e0 + lidx;
      const int   eec  = (ee < E) ? ee : (E - 1);
      const float vflg = (ee < E) ? 1.0f : 0.0f;
#pragma unroll
      for (int i = 0; i < 16; ++i) {
        const int k = kmap(i, hi);
        float v = 0.0f;
        if (k < 16) v = es[eec * 16 + k] * vflg;
        a[i] = (_Float16)v;
      }
    }

    // Layer 1: hidden(16x64) = A @ (fc1_w1/4)
    v8f acc[4];
#pragma unroll
    for (int t = 0; t < 4; ++t) { v8f z = {}; acc[t] = wmma_f16(a, B1[t], z); }

    // silu -> LDS as f16 in row-major [m][k] for re-layout into A operand
#pragma unroll
    for (int t = 0; t < 4; ++t)
#pragma unroll
      for (int r = 0; r < 8; ++r) {
        const int m = r + hi * 8;
        hb[m * 64 + t * 16 + lidx] = (_Float16)silu_f(acc[t][r]);
      }
    __builtin_amdgcn_wave_barrier();
    asm volatile("s_wait_dscnt 0" ::: "memory");

    // Layer 2: w(16x32) = hidden @ (fc1_w2/8), K=64 in two chunks
    v8f wacc[2];
#pragma unroll
    for (int t = 0; t < 2; ++t) {
      v8f z = {};
#pragma unroll
      for (int kc = 0; kc < 2; ++kc) {
        v16h ah;
#pragma unroll
        for (int i = 0; i < 16; ++i)
          ah[i] = hb[lidx * 64 + kc * 32 + kmap(i, hi)];
        z = wmma_f16(ah, B2[t][kc], z);
      }
      wacc[t] = z;
    }
    __builtin_amdgcn_wave_barrier();
    asm volatile("s_wait_dscnt 0" ::: "memory");

    // Gather xf[src] and scatter ef into mid[dst] (f32 atomics, L2-resident).
#pragma unroll
    for (int r = 0; r < 8; ++r) {
      const int   m   = r + hi * 8;
      const int   e   = e0 + m;
      const int   src = __shfl(m_src, m);
      const int   dst = __shfl(m_dst, m);
      const float sh0 = __shfl(m_sh0, m);
      const float s1x = __shfl(m_s1x, m);
      const float s1y = __shfl(m_s1y, m);
      const float s1z = __shfl(m_s1z, m);
      if (e < E) {
        const float xv = xf[src * 16 + lidx];
        const float w0 = wacc[0][r];
        const float w1 = wacc[1][r];
        atomicAdd(&mid[dst * 64 + lidx], w0 * xv * sh0);
        const float t1 = w1 * xv;
        atomicAdd(&mid[dst * 64 + 16 + lidx * 3 + 0], t1 * s1x);
        atomicAdd(&mid[dst * 64 + 16 + lidx * 3 + 1], t1 * s1y);
        atomicAdd(&mid[dst * 64 + 16 + lidx * 3 + 2], t1 * s1z);
      }
    }
  }
}

// ---------------------------------------------------------------------------
// Node nonlinearity between the two message-passing rounds.
// Produces sc2 (N x 8), y0 (N x 32), y1 (N x 24).
// ---------------------------------------------------------------------------
__global__ void node_mid_kernel(
    const float* __restrict__ mid,      // N x 64
    const float* __restrict__ x,        // N x 16 (node_features, for sc)
    const float* __restrict__ attr,
    const float* __restrict__ sc1_w,    // 16x40
    const float* __restrict__ lin2_w0,  // 16x40
    const float* __restrict__ lin2_w1,  // 16x8
    const float* __restrict__ lin3_w,   // 16x1
    const float* __restrict__ sc2_w,    // 32x8
    const float* __restrict__ lin1b_w0, // 32x32
    const float* __restrict__ lin1b_w1, // 8x8
    float* __restrict__ sc2o,           // N x 8
    float* __restrict__ y0o,            // N x 32
    float* __restrict__ y1o,            // N x 24
    int N) {
  const int i = blockIdx.x * blockDim.x + threadIdx.x;
  if (i >= N) return;
  const float inv_nn = 0.25f;                 // 1/sqrt(16)
  const float inv32  = 0.17677669529663687f;  // 1/sqrt(32)
  const float inv8   = 0.35355339059327373f;  // 1/sqrt(8)
  const float a = attr[i];

  float m0[16], m1[16][3];
#pragma unroll
  for (int u = 0; u < 16; ++u) m0[u] = mid[i * 64 + u] * inv_nn;
#pragma unroll
  for (int u = 0; u < 16; ++u)
#pragma unroll
    for (int c = 0; c < 3; ++c) m1[u][c] = mid[i * 64 + 16 + u * 3 + c] * inv_nn;

  float ang = 0.0f;
#pragma unroll
  for (int u = 0; u < 16; ++u) ang += m0[u] * lin3_w[u];
  ang *= 0.1f * 0.25f * a;
  const float cs = __cosf(ang), sn = __sinf(ang);

  float xs[16];
#pragma unroll
  for (int k = 0; k < 16; ++k) xs[k] = x[i * 16 + k];

  float y[64];
  for (int j = 0; j < 40; ++j) {
    float c0 = 0.0f, scv = 0.0f;
#pragma unroll
    for (int u = 0; u < 16; ++u) {
      c0  += m0[u] * lin2_w0[u * 40 + j];
      scv += xs[u] * sc1_w[u * 40 + j];
    }
    c0  *= 0.25f * a;
    scv *= 0.25f * a;
    y[j] = cs * scv + sn * c0;   // mask==1 region
  }
#pragma unroll
  for (int w = 0; w < 8; ++w)
#pragma unroll
    for (int c = 0; c < 3; ++c) {
      float s = 0.0f;
#pragma unroll
      for (int u = 0; u < 16; ++u) s += m1[u][c] * lin2_w1[u * 8 + w];
      y[40 + w * 3 + c] = s * 0.25f * a;   // mask==0: y = conv1 (sc part is 0)
    }

  float h0[32];
#pragma unroll
  for (int j = 0; j < 32; ++j) h0[j] = silu_f(y[j]);
  float g[8];
#pragma unroll
  for (int w = 0; w < 8; ++w) g[w] = sigmoid_f(y[32 + w]);
  float h1[8][3];
#pragma unroll
  for (int w = 0; w < 8; ++w)
#pragma unroll
    for (int c = 0; c < 3; ++c) h1[w][c] = y[40 + w * 3 + c] * g[w];

#pragma unroll
  for (int k = 0; k < 8; ++k) {
    float s = 0.0f;
#pragma unroll
    for (int j = 0; j < 32; ++j) s += h0[j] * sc2_w[j * 8 + k];
    sc2o[i * 8 + k] = s * inv32 * a;
  }
  for (int k = 0; k < 32; ++k) {
    float s = 0.0f;
#pragma unroll
    for (int j = 0; j < 32; ++j) s += h0[j] * lin1b_w0[j * 32 + k];
    y0o[i * 32 + k] = s * inv32 * a;
  }
#pragma unroll
  for (int w = 0; w < 8; ++w)
#pragma unroll
    for (int c = 0; c < 3; ++c) {
      float s = 0.0f;
#pragma unroll
      for (int u = 0; u < 8; ++u) s += h1[u][c] * lin1b_w1[u * 8 + w];
      y1o[i * 24 + w * 3 + c] = s * inv8 * a;
    }
}

// ---------------------------------------------------------------------------
// Edge MLP #2 on WMMA + gather y0/y1[src] + scatter-add into mid2 (N x 40).
// ---------------------------------------------------------------------------
__global__ __launch_bounds__(256) void edge_mlp2_kernel(
    const float* __restrict__ es,      // E x 16
    const float* __restrict__ eattr,   // E x 4
    const int*   __restrict__ esrc,
    const int*   __restrict__ edst,
    const float* __restrict__ fc2_w1,  // 16x64
    const float* __restrict__ fc2_w2,  // 64x40
    const float* __restrict__ y0,      // N x 32
    const float* __restrict__ y1,      // N x 24
    float* __restrict__ mid2,          // N x 40 (accumulated)
    int E) {
  __shared__ _Float16 hbuf[WAVES_PER_BLOCK * 16 * 64];
  const int lane = threadIdx.x & 31;
  const int wave = threadIdx.x >> 5;
  const int hi   = (lane >> 4) & 1;
  const int lidx = lane & 15;
  _Float16* hb = &hbuf[wave * 16 * 64];

  v16h B1[4], B2[3][2];
#pragma unroll
  for (int t = 0; t < 4; ++t) B1[t] = load_b(fc2_w1, 64, 16, t * 16, 0, 0.25f, lane);
#pragma unroll
  for (int t = 0; t < 3; ++t)
#pragma unroll
    for (int kc = 0; kc < 2; ++kc)
      B2[t][kc] = load_b(fc2_w2, 40, 64, t * 16, kc * 32, 0.125f, lane);

  const int ntiles = (E + 15) >> 4;
  const int stride = gridDim.x * WAVES_PER_BLOCK;
  for (int tile = blockIdx.x * WAVES_PER_BLOCK + wave; tile < ntiles; tile += stride) {
    const int e0 = tile << 4;

    const int  me    = e0 + lidx;
    const int  mec   = (me < E) ? me : (E - 1);
    const int  m_src = esrc[mec];
    const int  m_dst = edst[mec];
    const float m_sh0 = eattr[mec * 4 + 0];
    const float m_s1x = eattr[mec * 4 + 1];
    const float m_s1y = eattr[mec * 4 + 2];
    const float m_s1z = eattr[mec * 4 + 3];

    v16h a;
    {
      const int   ee   = e0 + lidx;
      const int   eec  = (ee < E) ? ee : (E - 1);
      const float vflg = (ee < E) ? 1.0f : 0.0f;
#pragma unroll
      for (int i = 0; i < 16; ++i) {
        const int k = kmap(i, hi);
        float v = 0.0f;
        if (k < 16) v = es[eec * 16 + k] * vflg;
        a[i] = (_Float16)v;
      }
    }

    v8f acc[4];
#pragma unroll
    for (int t = 0; t < 4; ++t) { v8f z = {}; acc[t] = wmma_f16(a, B1[t], z); }

#pragma unroll
    for (int t = 0; t < 4; ++t)
#pragma unroll
      for (int r = 0; r < 8; ++r) {
        const int m = r + hi * 8;
        hb[m * 64 + t * 16 + lidx] = (_Float16)silu_f(acc[t][r]);
      }
    __builtin_amdgcn_wave_barrier();
    asm volatile("s_wait_dscnt 0" ::: "memory");

    v8f wacc[3];
#pragma unroll
    for (int t = 0; t < 3; ++t) {
      v8f z = {};
#pragma unroll
      for (int kc = 0; kc < 2; ++kc) {
        v16h ah;
#pragma unroll
        for (int i = 0; i < 16; ++i)
          ah[i] = hb[lidx * 64 + kc * 32 + kmap(i, hi)];
        z = wmma_f16(ah, B2[t][kc], z);
      }
      wacc[t] = z;
    }
    __builtin_amdgcn_wave_barrier();
    asm volatile("s_wait_dscnt 0" ::: "memory");

#pragma unroll
    for (int r = 0; r < 8; ++r) {
      const int   m   = r + hi * 8;
      const int   e   = e0 + m;
      const int   src = __shfl(m_src, m);
      const int   dst = __shfl(m_dst, m);
      const float sh0 = __shfl(m_sh0, m);
      const float s1x = __shfl(m_s1x, m);
      const float s1y = __shfl(m_s1y, m);
      const float s1z = __shfl(m_s1z, m);
      if (e < E) {
        atomicAdd(&mid2[dst * 40 + lidx],
                  wacc[0][r] * y0[src * 32 + lidx] * sh0);
        atomicAdd(&mid2[dst * 40 + 16 + lidx],
                  wacc[1][r] * y0[src * 32 + 16 + lidx] * sh0);
        if (lidx < 8) {
          const float d = y1[src * 24 + lidx * 3 + 0] * s1x +
                          y1[src * 24 + lidx * 3 + 1] * s1y +
                          y1[src * 24 + lidx * 3 + 2] * s1z;
          atomicAdd(&mid2[dst * 40 + 32 + lidx],
                    wacc[2][r] * d * 0.5773502691896258f);  // 1/sqrt(3)
        }
      }
    }
  }
}

// ---------------------------------------------------------------------------
__global__ void node_out_kernel(
    const float* __restrict__ mid2,    // N x 40
    const float* __restrict__ attr,
    const float* __restrict__ sc2i,    // N x 8
    const float* __restrict__ lin2b_w, // 40x8
    const float* __restrict__ lin3b_w, // 40x1
    float* __restrict__ out,           // N x 8
    int N) {
  const int i = blockIdx.x * blockDim.x + threadIdx.x;
  if (i >= N) return;
  const float inv_nn = 0.25f;
  const float inv40  = 0.15811388300841897f;  // 1/sqrt(40)
  const float a = attr[i];

  float m[40];
#pragma unroll
  for (int j = 0; j < 40; ++j) m[j] = mid2[i * 40 + j] * inv_nn;

  float ang = 0.0f;
#pragma unroll
  for (int j = 0; j < 40; ++j) ang += m[j] * lin3b_w[j];
  ang *= 0.1f * inv40 * a;
  const float cs = __cosf(ang), sn = __sinf(ang);

#pragma unroll
  for (int k = 0; k < 8; ++k) {
    float s = 0.0f;
#pragma unroll
    for (int j = 0; j < 40; ++j) s += m[j] * lin2b_w[j * 8 + k];
    s *= inv40 * a;
    out[i * 8 + k] = cs * sc2i[i * 8 + k] + sn * s;
  }
}

// ---------------------------------------------------------------------------
extern "C" void kernel_launch(void* const* d_in, const int* in_sizes, int n_in,
                              void* d_out, int out_size, void* d_ws, size_t ws_size,
                              hipStream_t stream) {
  const float* node_features = (const float*)d_in[0];
  const float* node_attr     = (const float*)d_in[1];
  const float* edge_attr     = (const float*)d_in[2];
  const float* edge_scalars  = (const float*)d_in[3];
  const float* sc1_w         = (const float*)d_in[4];
  const float* lin1_w        = (const float*)d_in[5];
  const float* fc1_w1        = (const float*)d_in[6];
  const float* fc1_w2        = (const float*)d_in[7];
  const float* lin2_w0       = (const float*)d_in[8];
  const float* lin2_w1       = (const float*)d_in[9];
  const float* lin3_w        = (const float*)d_in[10];
  const float* sc2_w         = (const float*)d_in[11];
  const float* lin1b_w0      = (const float*)d_in[12];
  const float* lin1b_w1      = (const float*)d_in[13];
  const float* fc2_w1        = (const float*)d_in[14];
  const float* fc2_w2        = (const float*)d_in[15];
  const float* lin2b_w       = (const float*)d_in[16];
  const float* lin3b_w       = (const float*)d_in[17];
  const int*   edge_src      = (const int*)d_in[18];
  const int*   edge_dst      = (const int*)d_in[19];

  const int N = in_sizes[0] / 16;
  const int E = in_sizes[18];

  // Workspace layout (floats): xf | mid | sc2 | y0 | y1 | mid2  = N*184 floats
  float* ws   = (float*)d_ws;
  float* xf   = ws;
  float* mid  = xf  + (size_t)N * 16;
  float* sc2  = mid + (size_t)N * 64;
  float* y0   = sc2 + (size_t)N * 8;
  float* y1   = y0  + (size_t)N * 32;
  float* mid2 = y1  + (size_t)N * 24;

  const int TB = 256;

  zero_kernel<<<(N * 64 + TB - 1) / TB, TB, 0, stream>>>(mid,  N * 64);
  zero_kernel<<<(N * 40 + TB - 1) / TB, TB, 0, stream>>>(mid2, N * 40);

  node_pre_kernel<<<(N + TB - 1) / TB, TB, 0, stream>>>(
      node_features, node_attr, lin1_w, xf, N);

  const int ntiles = (E + 15) / 16;
  int eblocks = (ntiles + WAVES_PER_BLOCK - 1) / WAVES_PER_BLOCK;
  if (eblocks > 2048) eblocks = 2048;

  edge_mlp1_kernel<<<eblocks, TB, 0, stream>>>(
      edge_scalars, edge_attr, edge_src, edge_dst,
      fc1_w1, fc1_w2, xf, mid, E);

  node_mid_kernel<<<(N + TB - 1) / TB, TB, 0, stream>>>(
      mid, node_features, node_attr, sc1_w, lin2_w0, lin2_w1, lin3_w,
      sc2_w, lin1b_w0, lin1b_w1, sc2, y0, y1, N);

  edge_mlp2_kernel<<<eblocks, TB, 0, stream>>>(
      edge_scalars, edge_attr, edge_src, edge_dst,
      fc2_w1, fc2_w2, y0, y1, mid2, E);

  node_out_kernel<<<(N + TB - 1) / TB, TB, 0, stream>>>(
      mid2, node_attr, sc2, lin2b_w, lin3b_w, (float*)d_out, N);
}
